// MACETensorProductConv_34660386078944
// MI455X (gfx1250) — matrix operand order repaired and checked by reading, not compiled, not run on hardware
//
#include <hip/hip_runtime.h>
#include <hip/hip_bf16.h>

typedef __attribute__((ext_vector_type(16))) __bf16 v16bf;
typedef __attribute__((ext_vector_type(8)))  float  v8f;

#define C_DIM    128
#define E_TOT    100000
#define TILE_E   16
#define MLP_IN_D 64
#define MLP_HID  128
#define W_OUT    512
#define SH_DIM_D 16
#define N_DST_N  10000
#define OUT_COLS 2048

#define G1_KT 2   // 64/32
#define G1_NT 8   // 128/16
#define G2_KT 4   // 128/32
#define G2_NT 32  // 512/16
#define FRAG_ELEMS 512  // 32 lanes * 16 bf16

union FragU { v16bf v; uint4 q[2]; };

// ---------------------------------------------------------------------------
// Zero the output (harness poisons d_out with 0xAA before timing).
// ---------------------------------------------------------------------------
__global__ void zero_out_kernel(float4* __restrict__ out, int n4) {
    int i = blockIdx.x * blockDim.x + threadIdx.x;
    if (i < n4) out[i] = make_float4(0.f, 0.f, 0.f, 0.f);
}

// ---------------------------------------------------------------------------
// Pack W1*(1/sqrt(64)) and W2*(1/sqrt(128)) as bf16 in WMMA B-fragment order:
// ws[frag][lane][i], frag = ntile*KT + ktile, lane element i <-> K = (i>>3)*16 + (lane>>4)*8 + (i&7),
// N = lane&15 within the tile. Each lane's 16 bf16 are then contiguous (2x b128).
// ---------------------------------------------------------------------------
__global__ void pack_weights_kernel(const float* __restrict__ W1,
                                    const float* __restrict__ W2,
                                    __bf16* __restrict__ ws1,
                                    __bf16* __restrict__ ws2) {
    const int t = blockIdx.x * blockDim.x + threadIdx.x;
    const float s1 = 0.125f;                 // 1/sqrt(64)
    const float s2 = 0.08838834764831845f;   // 1/sqrt(128)
    const int n1 = G1_KT * G1_NT * FRAG_ELEMS;   // 8192
    const int n2 = G2_KT * G2_NT * FRAG_ELEMS;   // 65536
    if (t < n1) {
        int frag = t / FRAG_ELEMS, rem = t % FRAG_ELEMS;
        int lane = rem >> 4, i = rem & 15;
        int ktile = frag % G1_KT, ntile = frag / G1_KT;
        int k = ktile * 32 + (i >> 3) * 16 + (lane >> 4) * 8 + (i & 7);
        int n = ntile * 16 + (lane & 15);
        ws1[t] = (__bf16)(W1[k * MLP_HID + n] * s1);
    } else if (t - n1 < n2) {
        int t2 = t - n1;
        int frag = t2 / FRAG_ELEMS, rem = t2 % FRAG_ELEMS;
        int lane = rem >> 4, i = rem & 15;
        int ktile = frag % G2_KT, ntile = frag / G2_KT;
        int k = ktile * 32 + (i >> 3) * 16 + (lane >> 4) * 8 + (i & 7);
        int n = ntile * 16 + (lane & 15);
        ws2[t2] = (__bf16)(W2[k * W_OUT + n] * s2);
    }
}

// ---------------------------------------------------------------------------
// Fused: MLP (WMMA bf16, f32 acc, SiLU) -> tp_w ; outer product with gathered
// src features and spherical harmonics ; atomic scatter-add into node output.
// One workgroup = 16 edges, 128 threads = 4 waves; wave w owns l = w.
// ---------------------------------------------------------------------------
__global__ __launch_bounds__(128) void mace_tp_conv_kernel(
    const float* __restrict__ src_features,   // (N_SRC, 128)
    const float* __restrict__ edge_sh,        // (E, 16)
    const float* __restrict__ edge_emb,       // (E, 64)
    const int*   __restrict__ src,            // (E,)
    const int*   __restrict__ dst,            // (E,)
    const __bf16* __restrict__ ws1,           // packed W1 fragments
    const __bf16* __restrict__ ws2,           // packed W2 fragments
    float* __restrict__ out)                  // (N_DST, 2048), pre-zeroed
{
    __shared__ __bf16 s_emb[TILE_E][MLP_IN_D];   // A tile for GEMM1
    __shared__ __bf16 s_h[TILE_E][MLP_HID];      // SiLU(h), A tile for GEMM2
    __shared__ float  s_x[TILE_E][C_DIM];        // gathered src features
    __shared__ float  s_sh[TILE_E][SH_DIM_D];
    __shared__ int    s_dst[TILE_E];

    const int tid = threadIdx.x;
    const int e0  = blockIdx.x * TILE_E;

    // ---- cooperative staging into LDS ----
    for (int idx = tid; idx < TILE_E * MLP_IN_D; idx += 128) {
        int e = idx >> 6, k = idx & 63;
        s_emb[e][k] = (__bf16)edge_emb[(size_t)(e0 + e) * MLP_IN_D + k];
    }
    for (int idx = tid; idx < TILE_E * SH_DIM_D; idx += 128) {
        int e = idx >> 4, m = idx & 15;
        s_sh[e][m] = edge_sh[(size_t)(e0 + e) * SH_DIM_D + m];
    }
    if (tid < TILE_E) s_dst[tid] = dst[e0 + tid];
    for (int idx = tid; idx < TILE_E * C_DIM; idx += 128) {
        int e = idx >> 7, c = idx & 127;
        int sn = src[e0 + e];
        s_x[e][c] = src_features[(size_t)sn * C_DIM + c];
    }
    __syncthreads();

    const int lane = tid & 31;
    const int wave = tid >> 5;
    const int half = lane >> 4;
    const int row  = lane & 15;

    // ---- GEMM1: h = SiLU( emb(16x64) @ W1s(64x128) ), bf16 result to LDS ----
    {
        FragU a1[G1_KT];
#pragma unroll
        for (int kt = 0; kt < G1_KT; ++kt) {                 // hoist A fragments
            const int k0 = kt * 32 + half * 8;               // ISA 16-bit 16x32 A layout
            a1[kt].q[0] = *(const uint4*)&s_emb[row][k0];
            a1[kt].q[1] = *(const uint4*)&s_emb[row][k0 + 16];
        }
        for (int t = 0; t < 2; ++t) {
            const int nt = wave * 2 + t;
            v8f acc = {};
#pragma unroll
            for (int kt = 0; kt < G1_KT; ++kt) {
                FragU b;
                const uint4* bp = (const uint4*)(ws1 + ((size_t)(nt * G1_KT + kt) * 32 + lane) * 16);
                b.q[0] = bp[0];
                b.q[1] = bp[1];
                acc = __builtin_amdgcn_wmma_f32_16x16x32_bf16(
                    false, a1[kt].v, false, b.v, (short)0, acc, false, false);
            }
#pragma unroll
            for (int r = 0; r < 8; ++r) {                    // D: M = r + half*8, N = row
                float v = acc[r];
                // SiLU with fast reciprocal (v_rcp_f32) instead of IEEE divide
                v = v * __builtin_amdgcn_rcpf(1.0f + __expf(-v));
                s_h[r + half * 8][nt * 16 + row] = (__bf16)v;
            }
        }
    }
    __syncthreads();

    // ---- GEMM2: tp_w = h(16x128) @ W2s(128x512), fused outer-product scatter ----
    const int l        = wave;            // wave w handles angular momentum l = w
    const int d        = 2 * l + 1;       // 1,3,5,7
    const int sh_off   = l * l;           // 0,1,4,9
    const int col_base = C_DIM * sh_off;  // 0,128,512,1152

    FragU a2[G2_KT];
#pragma unroll
    for (int kt = 0; kt < G2_KT; ++kt) {                     // hoist A: same for all 8 N-tiles
        const int k0 = kt * 32 + half * 8;
        a2[kt].q[0] = *(const uint4*)&s_h[row][k0];
        a2[kt].q[1] = *(const uint4*)&s_h[row][k0 + 16];
    }

    for (int tt = 0; tt < 8; ++tt) {
        const int n2 = wave * 8 + tt;     // global N-tile 0..31
        v8f acc = {};
#pragma unroll
        for (int kt = 0; kt < G2_KT; ++kt) {
            FragU b;
            const uint4* bp = (const uint4*)(ws2 + ((size_t)(n2 * G2_KT + kt) * 32 + lane) * 16);
            b.q[0] = bp[0];
            b.q[1] = bp[1];
            acc = __builtin_amdgcn_wmma_f32_16x16x32_bf16(
                false, a2[kt].v, false, b.v, (short)0, acc, false, false);
        }
        const int c = tt * 16 + row;      // channel within 0..127
#pragma unroll
        for (int r = 0; r < 8; ++r) {
            const int erow = r + half * 8;               // edge row in tile
            const float wv = acc[r] * s_x[erow][c];      // tp_w * x_src
            float* op = out + (size_t)s_dst[erow] * OUT_COLS + col_base + c * d;
            for (int m = 0; m < d; ++m) {
                unsafeAtomicAdd(op + m, wv * s_sh[erow][sh_off + m]);
            }
        }
    }
}

// ---------------------------------------------------------------------------
extern "C" void kernel_launch(void* const* d_in, const int* in_sizes, int n_in,
                              void* d_out, int out_size, void* d_ws, size_t ws_size,
                              hipStream_t stream) {
    const float* src_features = (const float*)d_in[0];
    const float* edge_sh      = (const float*)d_in[1];
    const float* edge_emb     = (const float*)d_in[2];
    const float* W1           = (const float*)d_in[3];
    const float* W2           = (const float*)d_in[4];
    const int*   src          = (const int*)d_in[5];
    const int*   dst          = (const int*)d_in[6];
    float* out = (float*)d_out;

    __bf16* ws1 = (__bf16*)d_ws;                            // 8192 bf16
    __bf16* ws2 = ws1 + (size_t)G1_KT * G1_NT * FRAG_ELEMS; // 65536 bf16

    // 1) zero output
    const int n4 = (N_DST_N * OUT_COLS) / 4;
    zero_out_kernel<<<(n4 + 255) / 256, 256, 0, stream>>>((float4*)out, n4);

    // 2) convert+pack weights (bf16, WMMA B-fragment order, scales folded in)
    const int packThreads = (G1_KT * G1_NT + G2_KT * G2_NT) * FRAG_ELEMS; // 73728
    pack_weights_kernel<<<(packThreads + 255) / 256, 256, 0, stream>>>(W1, W2, ws1, ws2);

    // 3) fused MLP + tensor product + scatter
    mace_tp_conv_kernel<<<E_TOT / TILE_E, 128, 0, stream>>>(
        src_features, edge_sh, edge_emb, src, dst, ws1, ws2, out);
}
